// ColorHead_63007170232957
// MI455X (gfx1250) — compile-verified, matrix operand order checked by reading, not run on hardware
//
#include <hip/hip_runtime.h>
#include <hip/hip_bf16.h>

typedef __attribute__((ext_vector_type(16))) _Float16 v16h;
typedef __attribute__((ext_vector_type(8)))  float    v8f;

#define BATCH 2
#define CIN   3
#define C2    64
#define HW    512
#define HP    504            // mode-pool output spatial size
#define P0    (HP*HP)        // 254016 pixels per plane (divisible by 16)
#define PLANE (HW*HW)        // 262144

__device__ __forceinline__ float leakyf(float v)   { return v >= 0.f ? v : 0.01f*v; }
__device__ __forceinline__ float maxleaky(float v) { return v <= 0.1f ? v : 0.1f + 0.01f*(v-0.1f); }
__device__ __forceinline__ float minleaky(float v) { return v >= 0.1f ? v : 0.1f + 0.01f*(v-0.1f); }

// async global->LDS copy of one dword (CDNA5 path; per-lane, EXEC-masked)
__device__ __forceinline__ void async_copy_b32(void* lds_dst, const float* gsrc) {
  const unsigned ldsoff = (unsigned)(size_t)lds_dst;   // low 32 bits of generic LDS ptr = LDS offset
  asm volatile("global_load_async_to_lds_b32 %0, %1, off"
               :: "v"(ldsoff), "v"(gsrc) : "memory");
}
__device__ __forceinline__ void wait_async0() {
#if __has_builtin(__builtin_amdgcn_s_wait_asynccnt)
  __builtin_amdgcn_s_wait_asynccnt(0);
#else
  asm volatile("s_wait_asynccnt 0" ::: "memory");
#endif
}

// ---------------------------------------------------------------------------
// Kernel 1: quantize + 11x11 mode pool (17-bin histogram argmax), pad=1
// x tile staged via async global->LDS; OOB pre-zeroed (bin(0)==0 matches pad).
// grid: (32, 32, B*CIN), block (16,16)
// ---------------------------------------------------------------------------
__global__ void mode_pool_kernel(const float* __restrict__ x, float* __restrict__ h) {
  __shared__ float xt[26][28];
  __shared__ unsigned char bins[26][28];
  __shared__ int hist[256][17];                    // per-thread histograms
  const int plane = blockIdx.z;
  const float* xp = x + (size_t)plane * PLANE;
  const int tid = threadIdx.y * 16 + threadIdx.x;
  const int oy0 = blockIdx.y * 16, ox0 = blockIdx.x * 16;

  // stage 26x26 raw tile; padded coord (oy0-1 .. oy0+24)
  for (int q = tid; q < 26*26; q += 256) {
    const int ly = q / 26, lx = q % 26;
    const int iy = oy0 - 1 + ly, ix = ox0 - 1 + lx;
    if ((unsigned)iy < HW && (unsigned)ix < HW) {
      async_copy_b32(&xt[ly][lx], xp + iy*HW + ix);
    } else {
      xt[ly][lx] = 0.f;                            // pad value 0 -> bin 0
    }
  }
  wait_async0();
  __syncthreads();

  // quantize tile to bins
  for (int q = tid; q < 26*26; q += 256) {
    const int ly = q / 26, lx = q % 26;
    int bi = (int)rintf(xt[ly][lx] * (255.0f/16.0f));   // bin == round(x*255/16), clipped
    bi = bi < 0 ? 0 : (bi > 16 ? 16 : bi);
    bins[ly][lx] = (unsigned char)bi;
  }
  __syncthreads();

  int* hh = hist[tid];
  #pragma unroll
  for (int i = 0; i < 17; ++i) hh[i] = 0;
  #pragma unroll 1
  for (int dy = 0; dy < 11; ++dy)
    #pragma unroll
    for (int dx = 0; dx < 11; ++dx)
      hh[bins[threadIdx.y + dy][threadIdx.x + dx]]++;

  int best = 0, bc = hh[0];
  #pragma unroll
  for (int i = 1; i < 17; ++i) { if (hh[i] > bc) { bc = hh[i]; best = i; } }

  const int oy = oy0 + threadIdx.y, ox = ox0 + threadIdx.x;
  if (oy < HP && ox < HP)
    h[(size_t)plane * P0 + oy*HP + ox] = (float)best * (1.0f/16.0f);
}

// ---------------------------------------------------------------------------
// Kernel 2: WMMA head — 1x1 conv (3->64) + MaxLeaky + grouped 1x1 + MinLeaky
// GEMM: D[64 x Npix] = W1[64x3] * H[3 x Npix], K padded to 32, f16 inputs.
// Each wave: 16 pixels, 4 channel-tiles (v_wmma_f32_16x16x32_f16).
// grid: (ceil(P0/16/8), BATCH), block 256 (8 waves)
// ---------------------------------------------------------------------------
__global__ void head_kernel(const float* __restrict__ h,
                            const float* __restrict__ w1, const float* __restrict__ b1,
                            const float* __restrict__ w2, const float* __restrict__ b2,
                            float* __restrict__ d0) {
  const int wave = (blockIdx.x * blockDim.x + threadIdx.x) >> 5;
  const int lane = threadIdx.x & 31;
  const int b = blockIdx.y;
  const int pix = wave * 16 + (lane & 15);
  const bool valid = pix < P0;
  const int pc = valid ? pix : (P0 - 1);
  const bool lo16 = lane < 16;

  // B fragment: 32x16 f16; lanes 0-15 hold K=0..7/16..23, lanes 16-31 K=8..15/24..31 (zero here).
  v16h bf = {};
  if (lo16) {
    const float* hb = h + (size_t)b * CIN * P0;
    bf[0] = (_Float16)hb[0*(size_t)P0 + pc];
    bf[1] = (_Float16)hb[1*(size_t)P0 + pc];
    bf[2] = (_Float16)hb[2*(size_t)P0 + pc];
  }

  float* db = d0 + (size_t)b * C2 * P0;
  #pragma unroll 1
  for (int m = 0; m < 4; ++m) {
    // A fragment: 16x32 f16; lanes 0-15 hold M=lane, K=0..7/16..23; lanes 16-31 zero here
    v16h af = {};
    if (lo16) {
      const int ch = m * 16 + lane;
      af[0] = (_Float16)w1[ch*3 + 0];
      af[1] = (_Float16)w1[ch*3 + 1];
      af[2] = (_Float16)w1[ch*3 + 2];
    }
    v8f c = {};
    c = __builtin_amdgcn_wmma_f32_16x16x32_f16(false, af, false, bf, (short)0, c, false, false);
    // D layout: lane<16 -> (M=r, N=lane); lane>=16 -> (M=r+8, N=lane-16)
    const int chbase = m * 16 + (lo16 ? 0 : 8);
    if (valid) {
      #pragma unroll
      for (int j = 0; j < 4; ++j) {
        const int c0 = chbase + 2*j, c1 = c0 + 1;   // grouped pairs (2g, 2g+1) are lane-local
        float u0 = maxleaky(c[2*j]   + b1[c0]);
        float u1 = maxleaky(c[2*j+1] + b1[c1]);
        float o0 = w2[c0*2+0]*u0 + w2[c0*2+1]*u1 + b2[c0];
        float o1 = w2[c1*2+0]*u0 + w2[c1*2+1]*u1 + b2[c1];
        db[(size_t)c0 * P0 + pc] = minleaky(o0);
        db[(size_t)c1 * P0 + pc] = minleaky(o1);
      }
    }
  }
}

// ---------------------------------------------------------------------------
// Kernel 3: depthwise shared 3x3, stride 3, pad 1, + leaky (pyramid downgrade)
// ---------------------------------------------------------------------------
__global__ void downgrade_kernel(const float* __restrict__ in, float* __restrict__ out,
                                 int Hi, int Ho,
                                 const float* __restrict__ k9, const float* __restrict__ kb) {
  const int idx = blockIdx.x * blockDim.x + threadIdx.x;
  const int total = BATCH * C2 * Ho * Ho;
  if (idx >= total) return;
  const int ox = idx % Ho;
  int t = idx / Ho;
  const int oy = t % Ho;
  const int plane = t / Ho;
  const float* ip = in + (size_t)plane * Hi * Hi;
  float acc = kb[0];
  #pragma unroll
  for (int dy = 0; dy < 3; ++dy) {
    const int iy = 3*oy - 1 + dy;
    if ((unsigned)iy >= (unsigned)Hi) continue;
    #pragma unroll
    for (int dx = 0; dx < 3; ++dx) {
      const int ix = 3*ox - 1 + dx;
      if ((unsigned)ix >= (unsigned)Hi) continue;
      acc += k9[dy*3 + dx] * ip[iy*Hi + ix];
    }
  }
  out[idx] = leakyf(acc);
}

// ---------------------------------------------------------------------------
// Kernel 4: fused accumulation over all 5 pyramid levels:
//   score = sum_L leaky( BN( conv5x5_shared( bilinear_resize(d_L -> 512^2) ) ) )
// Per level, the 20x20 resized halo tile is materialized in LDS; score written once.
// grid: (32, 32, B*C2), block (16,16)
// ---------------------------------------------------------------------------
__global__ void accum_kernel(const float* __restrict__ dl1, const float* __restrict__ dl2,
                             const float* __restrict__ dl3, const float* __restrict__ dl4,
                             const float* __restrict__ dl5,
                             float* __restrict__ score,
                             const float* __restrict__ ik, const float* __restrict__ ibias,
                             const float* __restrict__ ig, const float* __restrict__ ibeta,
                             const float* __restrict__ imean, const float* __restrict__ ivar) {
  __shared__ float zt[20][20];
  __shared__ float kk[25];
  const int plane = blockIdx.z;
  const int c = plane % C2;
  const int tid = threadIdx.y * 16 + threadIdx.x;
  if (tid < 25) kk[tid] = ik[tid];

  const float scale = ig[c] * rsqrtf(ivar[c] + 1e-5f);
  const float mean = imean[c], beta = ibeta[c], cbias = ibias[0];

  const float* dls[5] = { dl1, dl2, dl3, dl4, dl5 };
  const int    S[5]   = { 168, 56, 19, 7, 3 };

  const int oy0 = blockIdx.y * 16, ox0 = blockIdx.x * 16;
  float acc = 0.f;

  #pragma unroll 1
  for (int L = 0; L < 5; ++L) {
    __syncthreads();                                   // zt reuse barrier
    const int s = S[L];
    const float* dp = dls[L] + (size_t)plane * s * s;
    const float sc = (float)s * (1.0f / 512.0f);
    for (int q = tid; q < 20*20; q += 256) {
      const int ly = q / 20, lx = q % 20;
      const int zy = oy0 + ly - 2, zx = ox0 + lx - 2;  // conv pad=2 -> zero outside 512^2
      float v = 0.f;
      if ((unsigned)zy < HW && (unsigned)zx < HW) {
        const float sy = ((float)zy + 0.5f) * sc - 0.5f;
        const float sx = ((float)zx + 0.5f) * sc - 0.5f;
        float fy0 = floorf(sy), fx0 = floorf(sx);
        const float wy = sy - fy0, wx = sx - fx0;
        int y0 = (int)fy0, x0 = (int)fx0;
        int y1 = y0 + 1, x1 = x0 + 1;
        y0 = y0 < 0 ? 0 : (y0 > s-1 ? s-1 : y0);
        y1 = y1 < 0 ? 0 : (y1 > s-1 ? s-1 : y1);
        x0 = x0 < 0 ? 0 : (x0 > s-1 ? s-1 : x0);
        x1 = x1 < 0 ? 0 : (x1 > s-1 ? s-1 : x1);
        const float v00 = dp[y0*s + x0], v01 = dp[y0*s + x1];
        const float v10 = dp[y1*s + x0], v11 = dp[y1*s + x1];
        v = (1.f-wy)*((1.f-wx)*v00 + wx*v01) + wy*((1.f-wx)*v10 + wx*v11);
      }
      zt[ly][lx] = v;
    }
    __syncthreads();
    float sum = cbias;
    #pragma unroll
    for (int dy = 0; dy < 5; ++dy)
      #pragma unroll
      for (int dx = 0; dx < 5; ++dx)
        sum += kk[dy*5 + dx] * zt[threadIdx.y + dy][threadIdx.x + dx];
    acc += leakyf((sum - mean) * scale + beta);
  }
  score[(size_t)plane * PLANE + (oy0 + threadIdx.y)*HW + (ox0 + threadIdx.x)] = acc;
}

// ---------------------------------------------------------------------------
// Kernel 5: one ft pass: t = in*A + B (affine = BN + shared 1x1), then
//   out = t + w_c * (t - boxsum11x11(t)/121)   (zero-padded SAME box sum)
// Tile staged via async global->LDS (raw), affine applied in-LDS, then
// separable box sum (row sums -> col sums).
// grid: (32, 32, B*C2), block (16,16)
// ---------------------------------------------------------------------------
__global__ void ft_kernel(const float* __restrict__ in, float* __restrict__ out,
                          const float* __restrict__ g, const float* __restrict__ be,
                          const float* __restrict__ me, const float* __restrict__ va,
                          const float* __restrict__ fk, const float* __restrict__ fb,
                          const float* __restrict__ ew) {
  __shared__ float t[26][26];
  __shared__ float rs[26][16];
  const int plane = blockIdx.z;
  const int c = plane % C2;
  const float sc = g[c] * rsqrtf(va[c] + 1e-5f);
  const float A = sc * fk[0];
  const float Bc = (be[c] - me[c]*sc) * fk[0] + fb[0];
  const float w = ew[c];
  const float* ip = in + (size_t)plane * PLANE;
  const int tid = threadIdx.y * 16 + threadIdx.x;
  const int oy0 = blockIdx.y * 16, ox0 = blockIdx.x * 16;

  // stage 26x26 raw tile via async copy; zero-pad outside the image
  for (int q = tid; q < 26*26; q += 256) {
    const int ly = q / 26, lx = q % 26;
    const int y = oy0 + ly - 5, x = ox0 + lx - 5;
    if ((unsigned)y < HW && (unsigned)x < HW) {
      async_copy_b32(&t[ly][lx], ip + y*HW + x);
    } else {
      t[ly][lx] = 0.f;
    }
  }
  wait_async0();
  __syncthreads();

  // apply affine in LDS (keep zero padding intact)
  for (int q = tid; q < 26*26; q += 256) {
    const int ly = q / 26, lx = q % 26;
    const int y = oy0 + ly - 5, x = ox0 + lx - 5;
    if ((unsigned)y < HW && (unsigned)x < HW)
      t[ly][lx] = t[ly][lx] * A + Bc;
  }
  __syncthreads();

  for (int q = tid; q < 26*16; q += 256) {
    const int ly = q / 16, lx = q % 16;
    float s = 0.f;
    #pragma unroll
    for (int dx = 0; dx < 11; ++dx) s += t[ly][lx + dx];
    rs[ly][lx] = s;
  }
  __syncthreads();
  float s = 0.f;
  #pragma unroll
  for (int dy = 0; dy < 11; ++dy) s += rs[threadIdx.y + dy][threadIdx.x];
  const float tv = t[threadIdx.y + 5][threadIdx.x + 5];
  out[(size_t)plane * PLANE + (oy0 + threadIdx.y)*HW + (ox0 + threadIdx.x)] =
      tv + w * (tv - s * (1.0f/121.0f));
}

// ---------------------------------------------------------------------------
extern "C" void kernel_launch(void* const* d_in, const int* in_sizes, int n_in,
                              void* d_out, int out_size, void* d_ws, size_t ws_size,
                              hipStream_t stream) {
  (void)in_sizes; (void)n_in; (void)out_size; (void)ws_size;
  const float* x        = (const float*)d_in[0];
  const float* w1       = (const float*)d_in[1];
  const float* b1       = (const float*)d_in[2];
  const float* w2       = (const float*)d_in[3];
  const float* b2       = (const float*)d_in[4];
  const float* down_k   = (const float*)d_in[5];
  const float* down_b   = (const float*)d_in[6];
  const float* ft_gamma = (const float*)d_in[7];
  const float* ft_beta  = (const float*)d_in[8];
  const float* ft_mean  = (const float*)d_in[9];
  const float* ft_var   = (const float*)d_in[10];
  const float* ft_k     = (const float*)d_in[11];
  const float* ft_b     = (const float*)d_in[12];
  const float* emph_w   = (const float*)d_in[13];
  const float* interp_k = (const float*)d_in[14];
  const float* interp_b = (const float*)d_in[15];
  const float* i_gamma  = (const float*)d_in[16];
  const float* i_beta   = (const float*)d_in[17];
  const float* i_mean   = (const float*)d_in[18];
  const float* i_var    = (const float*)d_in[19];
  float* outp = (float*)d_out;

  // workspace layout
  char* ws = (char*)d_ws;
  size_t off = 0;
  auto take = [&](size_t bytes) { char* p = ws + off; off += (bytes + 255) & ~(size_t)255; return (float*)p; };
  float* hbuf   = take((size_t)BATCH*CIN*P0*4);             // 6.1 MB
  float* scoreA = take((size_t)BATCH*C2*PLANE*4);           // 134 MB; overlays d0 (130 MB), d0 dead before accum
  float* d0     = scoreA;
  float* dlv[5];
  const int S[6] = { HP, 168, 56, 19, 7, 3 };
  for (int L = 0; L < 5; ++L) dlv[L] = take((size_t)BATCH*C2*S[L+1]*S[L+1]*4);   // 16.3 MB total

  // 1) mode pool
  mode_pool_kernel<<<dim3(32, 32, BATCH*CIN), dim3(16, 16), 0, stream>>>(x, hbuf);

  // 2) WMMA head -> d0 [B,64,504,504]
  {
    const int waves = P0 / 16;                  // 15876, exact
    const int blocks = (waves + 7) / 8;
    head_kernel<<<dim3(blocks, BATCH), 256, 0, stream>>>(hbuf, w1, b1, w2, b2, d0);
  }

  // 3) pyramid downgrades d0 -> d1..d5
  {
    const float* src = d0;
    for (int L = 0; L < 5; ++L) {
      const int Hi = S[L], Ho = S[L+1];
      const int total = BATCH * C2 * Ho * Ho;
      downgrade_kernel<<<(total + 255)/256, 256, 0, stream>>>(src, dlv[L], Hi, Ho, down_k, down_b);
      src = dlv[L];
    }
  }

  // 4) fused resize+conv5+BN+leaky accumulation over all 5 levels -> scoreA
  accum_kernel<<<dim3(32, 32, BATCH*C2), dim3(16, 16), 0, stream>>>(
      dlv[0], dlv[1], dlv[2], dlv[3], dlv[4], scoreA,
      interp_k, interp_b, i_gamma, i_beta, i_mean, i_var);

  // 5) three ft passes (ping-pong so pass 3 ends in d_out)
  ft_kernel<<<dim3(32, 32, BATCH*C2), dim3(16, 16), 0, stream>>>(
      scoreA, outp, ft_gamma, ft_beta, ft_mean, ft_var, ft_k, ft_b, emph_w);
  ft_kernel<<<dim3(32, 32, BATCH*C2), dim3(16, 16), 0, stream>>>(
      outp, scoreA, ft_gamma, ft_beta, ft_mean, ft_var, ft_k, ft_b, emph_w);
  ft_kernel<<<dim3(32, 32, BATCH*C2), dim3(16, 16), 0, stream>>>(
      scoreA, outp, ft_gamma, ft_beta, ft_mean, ft_var, ft_k, ft_b, emph_w);
}
